// SNNLoss_47622597378740
// MI455X (gfx1250) — compile-verified, matrix operand order checked by reading, not run on hardware
//
#include <hip/hip_runtime.h>
#include <hip/hip_bf16.h>

#define BB 4096
#define DD 128
#define NN 8192
#define CSPLIT 8
#define COLS_PER_CHUNK (NN / CSPLIT)
#define TILES_PER_CHUNK (COLS_PER_CHUNK / 16)
#define LDS_STRIDE 272            // 136 halves * 2B: 16B-aligned, de-aligns banks
#define TILE_BYTES (16 * LDS_STRIDE)

typedef __attribute__((ext_vector_type(16))) _Float16 v16h;
typedef __attribute__((ext_vector_type(8)))  float    v8f;

union Frag  { v16h v; uint4 q[2]; };
union Pack8 { _Float16 h[8]; uint4 q; };

#define NLOG2E (-1.44269504088896340736f)
#define LN2    (0.69314718055994530942f)

__device__ __forceinline__ const float* zrow(const float* x, const float* y, int i) {
  return (i < BB) ? (x + (size_t)i * DD) : (y + (size_t)(i - BB) * DD);
}

// Kernel 1: per-row squared norms, exact positive-pair sim, zero accumulators,
// optional f16 copy of z for the async-staged main kernel.
__global__ void snn_prep(const float* __restrict__ x, const float* __restrict__ y,
                         float* __restrict__ sq, float* __restrict__ pos,
                         float* __restrict__ rowS, _Float16* __restrict__ zh,
                         float* __restrict__ out) {
  const int i = blockIdx.x;
  const int t = threadIdx.x;
  const float* zr = zrow(x, y, i);
  const float* zp = (i < BB) ? (y + (size_t)i * DD) : (x + (size_t)(i - BB) * DD);
  float a = zr[t];
  float b = zp[t];
  if (zh) zh[(size_t)i * DD + t] = (_Float16)a;
  __shared__ float r1[DD];
  __shared__ float r2[DD];
  r1[t] = a * a;
  float d = a - b;
  r2[t] = d * d;
  __syncthreads();
  for (int off = DD / 2; off > 0; off >>= 1) {
    if (t < off) { r1[t] += r1[t + off]; r2[t] += r2[t + off]; }
    __syncthreads();
  }
  if (t == 0) {
    sq[i]   = r1[0];
    pos[i]  = -__builtin_sqrtf(fmaxf(r2[0], 0.0f));  // sim of the unique positive
    rowS[i] = 0.0f;
    if (i == 0) out[0] = 0.0f;
  }
}

// Shared epilogue: d2 -> exp(-sqrt(d2)) with raw v_sqrt_f32 / v_exp_f32.
__device__ __forceinline__ float expnegdist(float d2) {
  d2 = fmaxf(d2, 0.0f);
  return __builtin_amdgcn_exp2f(NLOG2E * __builtin_amdgcn_sqrtf(d2));
}

// Kernel 2a (preferred): fused Gram + exp-sum sweep with ASYNCcnt-tracked
// global->LDS tile staging (double buffered, one barrier per tile) and
// v_wmma_f32_16x16x32_f16 for the dot products.
__global__ void __launch_bounds__(256)
snn_main_async(const _Float16* __restrict__ zh, const float* __restrict__ sq,
               float* __restrict__ rowS) {
  __shared__ __align__(16) unsigned char lds[2 * TILE_BYTES];

  const int tid  = threadIdx.x;
  const int wave = tid >> 5;
  const int lane = tid & 31;
  const int m    = lane & 15;   // row (A) / col (B,C,D) index within tile
  const int h    = lane >> 4;   // lane half
  const int row_base   = blockIdx.x * 128 + wave * 16;
  const int row0       = row_base + 8 * h;        // C/D: VGPR r -> row row0+r
  const int chunk_base = blockIdx.y * COLS_PER_CHUNK;

  // A fragments: 16 rows x K=128, straight f16 loads (ISA 16-bit A 16x32 layout).
  Frag a[4];
  {
    const uint4* ar = (const uint4*)(zh + (size_t)(row_base + m) * DD);
    #pragma unroll
    for (int kk = 0; kk < 4; ++kk) {
      a[kk].q[0] = ar[kk * 4 + h];       // K = 32kk + 8h .. +8
      a[kk].q[1] = ar[kk * 4 + 2 + h];   // K = 32kk + 16 + 8h .. +8
    }
  }

  float sqr[8];
  #pragma unroll
  for (int r = 0; r < 8; ++r) sqr[r] = sq[row0 + r];
  float acc[8];
  #pragma unroll
  for (int r = 0; r < 8; ++r) acc[r] = 0.0f;

  // Cooperative async loader: 256 threads x b128 == one 16x128 f16 tile.
  const int trow = tid >> 4;
  const int tseg = tid & 15;
  const unsigned lds_off0 =
      (unsigned)(uintptr_t)&lds[trow * LDS_STRIDE + tseg * 16];
  unsigned long long gaddr =
      (unsigned long long)(uintptr_t)(zh + (size_t)(chunk_base + trow) * DD + tseg * 8);

  // Prologue: stage tile 0 into buffer 0.
  asm volatile("global_load_async_to_lds_b128 %0, %1, off"
               :: "v"(lds_off0), "v"(gaddr) : "memory");
  gaddr += 16ull * DD * sizeof(_Float16);  // next tile (16 rows)

  for (int t = 0; t < TILES_PER_CHUNK; ++t) {
    asm volatile("s_wait_asynccnt 0x0" ::: "memory");  // my tile-t bytes landed
    __syncthreads();  // everyone's tile-t landed AND tile-(t-1) compute done

    if (t + 1 < TILES_PER_CHUNK) {  // overlap: stage tile t+1 into other buffer
      unsigned dst = lds_off0 + (unsigned)(((t + 1) & 1) * TILE_BYTES);
      asm volatile("global_load_async_to_lds_b128 %0, %1, off"
                   :: "v"(dst), "v"(gaddr) : "memory");
      gaddr += 16ull * DD * sizeof(_Float16);
    }

    const int col_base = chunk_base + t * 16;
    const float sqc = sq[col_base + m];
    const unsigned char* buf = &lds[(t & 1) * TILE_BYTES];

    v8f c = {};
    #pragma unroll
    for (int kk = 0; kk < 4; ++kk) {
      Frag b;  // Gram matrix: B fragment = A-style fragment of the column rows
      const unsigned char* bp = buf + m * LDS_STRIDE + (kk * 32 + 8 * h) * 2;
      b.q[0] = *(const uint4*)(bp);
      b.q[1] = *(const uint4*)(bp + 32);
      c = __builtin_amdgcn_wmma_f32_16x16x32_f16(false, a[kk].v, false, b.v,
                                                 (short)0, c, false, false);
    }

    if (row_base == col_base) {        // only the matching tile holds a diagonal
      const int j = col_base + m;
      #pragma unroll
      for (int r = 0; r < 8; ++r) {
        float e = expnegdist(sqr[r] + sqc - 2.0f * c[r]);
        if (row0 + r == j) e = 0.0f;
        acc[r] += e;
      }
    } else {
      #pragma unroll
      for (int r = 0; r < 8; ++r)
        acc[r] += expnegdist(sqr[r] + sqc - 2.0f * c[r]);
    }
  }

  #pragma unroll
  for (int r = 0; r < 8; ++r) {
    float v = acc[r];
    v += __shfl_xor(v, 1, 32);
    v += __shfl_xor(v, 2, 32);
    v += __shfl_xor(v, 4, 32);
    v += __shfl_xor(v, 8, 32);
    if (m == 0) atomicAdd(&rowS[row0 + r], v);
  }
}

// Kernel 2b (fallback if ws has no room for the f16 copy): convert-in-flight
// staging through VGPRs (same math, synchronous ds_store path).
__global__ void __launch_bounds__(256)
snn_main_conv(const float* __restrict__ x, const float* __restrict__ y,
              const float* __restrict__ sq, float* __restrict__ rowS) {
  __shared__ __align__(16) unsigned char lds[TILE_BYTES];

  const int tid  = threadIdx.x;
  const int wave = tid >> 5;
  const int lane = tid & 31;
  const int m    = lane & 15;
  const int h    = lane >> 4;
  const int row_base   = blockIdx.x * 128 + wave * 16;
  const int row0       = row_base + 8 * h;
  const int chunk_base = blockIdx.y * COLS_PER_CHUNK;

  Frag a[4];
  {
    const float* ar = zrow(x, y, row_base + m);
    for (int kk = 0; kk < 4; ++kk) {
      Pack8 p0, p1;
      const float* s0 = ar + kk * 32 + 8 * h;
      const float* s1 = ar + kk * 32 + 16 + 8 * h;
      #pragma unroll
      for (int e = 0; e < 8; ++e) { p0.h[e] = (_Float16)s0[e]; p1.h[e] = (_Float16)s1[e]; }
      a[kk].q[0] = p0.q;
      a[kk].q[1] = p1.q;
    }
  }

  float sqr[8];
  #pragma unroll
  for (int r = 0; r < 8; ++r) sqr[r] = sq[row0 + r];
  float acc[8];
  #pragma unroll
  for (int r = 0; r < 8; ++r) acc[r] = 0.0f;

  const int trow = tid >> 4;
  const int tseg = tid & 15;

  for (int t = 0; t < TILES_PER_CHUNK; ++t) {
    const int col_base = chunk_base + t * 16;
    const float* src = zrow(x, y, col_base + trow) + tseg * 8;
    float4 f0 = *(const float4*)(src);
    float4 f1 = *(const float4*)(src + 4);
    Pack8 p;
    p.h[0] = (_Float16)f0.x; p.h[1] = (_Float16)f0.y;
    p.h[2] = (_Float16)f0.z; p.h[3] = (_Float16)f0.w;
    p.h[4] = (_Float16)f1.x; p.h[5] = (_Float16)f1.y;
    p.h[6] = (_Float16)f1.z; p.h[7] = (_Float16)f1.w;
    __syncthreads();
    *(uint4*)(lds + trow * LDS_STRIDE + tseg * 16) = p.q;
    __syncthreads();

    const float sqc = sq[col_base + m];
    v8f c = {};
    #pragma unroll
    for (int kk = 0; kk < 4; ++kk) {
      Frag b;
      const unsigned char* bp = lds + m * LDS_STRIDE + (kk * 32 + 8 * h) * 2;
      b.q[0] = *(const uint4*)(bp);
      b.q[1] = *(const uint4*)(bp + 32);
      c = __builtin_amdgcn_wmma_f32_16x16x32_f16(false, a[kk].v, false, b.v,
                                                 (short)0, c, false, false);
    }

    if (row_base == col_base) {
      const int j = col_base + m;
      #pragma unroll
      for (int r = 0; r < 8; ++r) {
        float e = expnegdist(sqr[r] + sqc - 2.0f * c[r]);
        if (row0 + r == j) e = 0.0f;
        acc[r] += e;
      }
    } else {
      #pragma unroll
      for (int r = 0; r < 8; ++r)
        acc[r] += expnegdist(sqr[r] + sqc - 2.0f * c[r]);
    }
  }

  #pragma unroll
  for (int r = 0; r < 8; ++r) {
    float v = acc[r];
    v += __shfl_xor(v, 1, 32);
    v += __shfl_xor(v, 2, 32);
    v += __shfl_xor(v, 4, 32);
    v += __shfl_xor(v, 8, 32);
    if (m == 0) atomicAdd(&rowS[row0 + r], v);
  }
}

// Kernel 3: per-row loss and mean.
__global__ void snn_final(const float* __restrict__ pos, const float* __restrict__ rowS,
                          float* __restrict__ out) {
  const int i = blockIdx.x * 256 + threadIdx.x;
  float num = __builtin_amdgcn_exp2f(-NLOG2E * pos[i]);          // exp(pos)
  float arg = num / rowS[i] + 1.17549435e-38f;                   // + f32 tiny
  float li  = -LN2 * __builtin_amdgcn_logf(arg);                 // -ln(arg)
  __shared__ float red[256];
  red[threadIdx.x] = li;
  __syncthreads();
  for (int off = 128; off > 0; off >>= 1) {
    if (threadIdx.x < off) red[threadIdx.x] += red[threadIdx.x + off];
    __syncthreads();
  }
  if (threadIdx.x == 0) atomicAdd(out, red[0] * (1.0f / NN));
}

extern "C" void kernel_launch(void* const* d_in, const int* in_sizes, int n_in,
                              void* d_out, int out_size, void* d_ws, size_t ws_size,
                              hipStream_t stream) {
  const float* x = (const float*)d_in[0];
  const float* y = (const float*)d_in[1];
  float* out  = (float*)d_out;
  float* sq   = (float*)d_ws;          // [NN]
  float* pos  = sq + NN;               // [NN]
  float* rowS = pos + NN;              // [NN]
  const size_t base_bytes = (size_t)3 * NN * sizeof(float);      // 96 KB
  const size_t zh_bytes   = (size_t)NN * DD * sizeof(_Float16);  // 2 MB
  const bool use_async = ws_size >= base_bytes + zh_bytes;
  _Float16* zh = use_async ? (_Float16*)((char*)d_ws + base_bytes) : nullptr;

  snn_prep<<<NN, DD, 0, stream>>>(x, y, sq, pos, rowS, zh, out);
  dim3 grid(NN / 128, CSPLIT);
  if (use_async) {
    snn_main_async<<<grid, 256, 0, stream>>>(zh, sq, rowS);
  } else {
    snn_main_conv<<<grid, 256, 0, stream>>>(x, y, sq, rowS);
  }
  snn_final<<<NN / 256, 256, 0, stream>>>(pos, rowS, out);
}
